// StoWinMSA_69707319214574
// MI455X (gfx1250) — matrix operand order repaired and checked by reading, not compiled, not run on hardware
//
#include <hip/hip_runtime.h>

typedef _Float16 h16;
typedef __attribute__((ext_vector_type(16))) _Float16 v16h;
typedef __attribute__((ext_vector_type(8)))  _Float16 v8h;
typedef __attribute__((ext_vector_type(8)))  float    v8f;

constexpr int kC  = 384;   // channels
constexpr int kNH = 12;    // heads
constexpr int kD  = 32;    // head dim
constexpr int kL  = 64;    // tokens per 8x8 window
constexpr int kH  = 256;
constexpr int kW  = 256;
constexpr int k3C = 1152;

__device__ __forceinline__ v8f zero8() {
  v8f z;
#pragma unroll
  for (int i = 0; i < 8; ++i) z[i] = 0.f;
  return z;
}

__device__ __forceinline__ v8f wmma_f16(v16h a, v16h b, v8f c) {
  return __builtin_amdgcn_wmma_f32_16x16x32_f16(false, a, false, b, (short)0, c, false, false);
}

// Build a 16x32 f16 A-fragment from two 8-half contiguous runs (K order used here:
// runs at K = 16*half and K = 16*half + 8 for each lane's row).
__device__ __forceinline__ v16h mkA(const h16* p0, const h16* p1) {
  v8h lo = *(const v8h*)p0;
  v8h hi = *(const v8h*)p1;
  return __builtin_shufflevector(lo, hi, 0, 1, 2, 3, 4, 5, 6, 7,
                                 8, 9, 10, 11, 12, 13, 14, 15);
}

// ---------------------------------------------------------------------------
// One workgroup == one 8x8 window of one shift.  384 threads = 12 waves.
// ---------------------------------------------------------------------------
__global__ __launch_bounds__(384)
void win_msa_kernel(const float* __restrict__ x,
                    const float* __restrict__ b_qkv,
                    const float* __restrict__ b_proj,
                    const h16* __restrict__ wqkvT,   // [1152][384] f16
                    const h16* __restrict__ wprojT,  // [384][384]  f16
                    float* __restrict__ out,
                    int sh, int sw, int nh, int nw, int first) {
  // LDS: 48KB + 48KB + 48KB + 12KB = 156KB
  __shared__ __attribute__((aligned(32))) h16 qbuf[kNH * kL * kD]; // [head][tok][d]; reused for attn out
  __shared__ __attribute__((aligned(32))) h16 kbuf[kNH * kL * kD]; // [head][tok][d]
  __shared__ __attribute__((aligned(32))) h16 vbuf[kNH * kD * kL]; // [head][d][tok] (transposed)
  __shared__ __attribute__((aligned(32))) h16 abuf[kNH * 16 * kD]; // per-wave prob staging [wave][16][32]

  const int tid  = threadIdx.x;
  const int wv   = tid >> 5;       // wave id 0..11
  const int lane = tid & 31;
  const int sel  = lane & 15;      // column / row selector
  const int hlf  = lane >> 4;      // lane half

  const int nWin = nh * nw;
  const int b    = blockIdx.x / nWin;
  const int rem  = blockIdx.x - b * nWin;
  const int whi  = rem / nw;
  const int wwi  = rem - whi * nw;

  const float scale = 0.17677669529663687f; // 32^-0.5

  // Per-M-tile source row pointers (token = mt*16 + sel) with zero-padding
  // mask (tokens outside the image contribute zero A rows => qkv = bias,
  // exactly matching the reference's jnp.pad).
  const float* rowp[4];
  int rowok[4];
#pragma unroll
  for (int mt = 0; mt < 4; ++mt) {
    const int t  = mt * 16 + sel;
    const int hh = whi * 8 + (t >> 3) - sh;
    const int ww = wwi * 8 + (t & 7) - sw;
    const int ok = (hh >= 0) & (hh < kH) & (ww >= 0) & (ww < kW);
    rowok[mt] = ok;
    rowp[mt]  = ok ? (x + (((size_t)b * kH + hh) * kW + ww) * (size_t)kC) : x;
  }

  // ---------------- Phase 1: qkv = xw @ Wqkv + b ----------------
  // 72 N-tiles x 4 M-tiles; wave w owns N-tiles [6w, 6w+6), processed as
  // 3 pairs so each global A-fragment feeds two WMMAs.
  for (int g = 0; g < 3; ++g) {
    const int j0a = (wv * 6 + g * 2) * 16;  // first column tile of the pair
    const int j0b = j0a + 16;
    v8f acc0[4], acc1[4];
#pragma unroll
    for (int mt = 0; mt < 4; ++mt) { acc0[mt] = zero8(); acc1[mt] = zero8(); }

    for (int k = 0; k < 12; ++k) {
      const int k0 = k * 32;
      // B fragments: 16 contiguous f16 at WqkvT[j+sel][k0 + 16*hlf]
      const v16h bf0 = *(const v16h*)(wqkvT + (size_t)(j0a + sel) * kC + k0 + 16 * hlf);
      const v16h bf1 = *(const v16h*)(wqkvT + (size_t)(j0b + sel) * kC + k0 + 16 * hlf);
#pragma unroll
      for (int mt = 0; mt < 4; ++mt) {
        v16h af;
        if (rowok[mt]) {
          const float* ap = rowp[mt] + k0 + 16 * hlf;
          const float4 f0 = *(const float4*)(ap);
          const float4 f1 = *(const float4*)(ap + 4);
          const float4 f2 = *(const float4*)(ap + 8);
          const float4 f3 = *(const float4*)(ap + 12);
          af[0] = (h16)f0.x; af[1] = (h16)f0.y; af[2]  = (h16)f0.z; af[3]  = (h16)f0.w;
          af[4] = (h16)f1.x; af[5] = (h16)f1.y; af[6]  = (h16)f1.z; af[7]  = (h16)f1.w;
          af[8] = (h16)f2.x; af[9] = (h16)f2.y; af[10] = (h16)f2.z; af[11] = (h16)f2.w;
          af[12] = (h16)f3.x; af[13] = (h16)f3.y; af[14] = (h16)f3.z; af[15] = (h16)f3.w;
        } else {
#pragma unroll
          for (int i = 0; i < 16; ++i) af[i] = (h16)0.f;
        }
        acc0[mt] = wmma_f16(af, bf0, acc0[mt]);
        acc1[mt] = wmma_f16(af, bf1, acc1[mt]);
      }
    }

    // Scatter both tiles to q/k/v LDS (each tile spans one head, one of q/k/v).
#pragma unroll
    for (int p = 0; p < 2; ++p) {
      const int j0    = p ? j0b : j0a;
      const int which = j0 / kC;            // 0=q 1=k 2=v
      const int head  = (j0 % kC) / kD;
      const int dim   = (j0 % kD) + sel;    // j0%kD is 0 or 16
      const float bj  = b_qkv[j0 + sel];
#pragma unroll
      for (int mt = 0; mt < 4; ++mt) {
#pragma unroll
        for (int r = 0; r < 8; ++r) {
          const int tok = mt * 16 + r + 8 * hlf;
          const float v = (p ? acc1[mt][r] : acc0[mt][r]) + bj;
          if (which == 0)      qbuf[(head * kL + tok) * kD + dim] = (h16)(v * scale);
          else if (which == 1) kbuf[(head * kL + tok) * kD + dim] = (h16)v;
          else                 vbuf[(head * kD + dim) * kL + tok] = (h16)v;
        }
      }
    }
  }
  __syncthreads();

  // ---------------- Phase 2: per-head attention (head = wave id) ----------
  {
    const h16* qh = qbuf + wv * kL * kD;
    const h16* kh = kbuf + wv * kL * kD;
    const h16* vh = vbuf + wv * kD * kL;
    h16*       ab = abuf + wv * 16 * kD;

    for (int mt = 0; mt < 4; ++mt) {
      // scores = q @ k^T (4 N-tiles, K = d = 32 in one step)
      const h16* qp = qh + (mt * 16 + sel) * kD;
      const v16h qa = mkA(qp + 16 * hlf, qp + 16 * hlf + 8);
      v8f s[4];
#pragma unroll
      for (int nt = 0; nt < 4; ++nt) {
        const v16h kb = *(const v16h*)(kh + (nt * 16 + sel) * kD + 16 * hlf);
        s[nt] = wmma_f16(qa, kb, zero8());
      }

      // row softmax (row r+8*hlf lives in acc element r across lanes of one half;
      // xor masks <= 8 never cross the 16-lane halves)
      float rinv[8];
#pragma unroll
      for (int r = 0; r < 8; ++r) {
        float m = fmaxf(fmaxf(s[0][r], s[1][r]), fmaxf(s[2][r], s[3][r]));
        m = fmaxf(m, __shfl_xor(m, 1, 32));
        m = fmaxf(m, __shfl_xor(m, 2, 32));
        m = fmaxf(m, __shfl_xor(m, 4, 32));
        m = fmaxf(m, __shfl_xor(m, 8, 32));
        float sum = 0.f;
#pragma unroll
        for (int nt = 0; nt < 4; ++nt) {
          const float e = __expf(s[nt][r] - m);
          s[nt][r] = e;
          sum += e;
        }
        sum += __shfl_xor(sum, 1, 32);
        sum += __shfl_xor(sum, 2, 32);
        sum += __shfl_xor(sum, 4, 32);
        sum += __shfl_xor(sum, 8, 32);
        rinv[r] = 1.f / sum;
      }

      // out = attn @ v, two K-chunks of 32 via per-wave LDS staging
      v8f oacc[2] = {zero8(), zero8()};
      for (int c = 0; c < 2; ++c) {
#pragma unroll
        for (int q2 = 0; q2 < 2; ++q2) {
          const int nt = 2 * c + q2;
#pragma unroll
          for (int r = 0; r < 8; ++r)
            ab[(r + 8 * hlf) * kD + q2 * 16 + sel] = (h16)(s[nt][r] * rinv[r]);
        }
        asm volatile("s_wait_dscnt 0" ::: "memory");
        const h16* pp = ab + sel * kD;
        const v16h pa = mkA(pp + 16 * hlf, pp + 16 * hlf + 8);
#pragma unroll
        for (int dv = 0; dv < 2; ++dv) {
          const v16h vb = *(const v16h*)(vh + (dv * 16 + sel) * kL + c * 32 + 16 * hlf);
          oacc[dv] = wmma_f16(pa, vb, oacc[dv]);
        }
      }

      // store head output over the (now dead) q rows of this M-tile
#pragma unroll
      for (int dv = 0; dv < 2; ++dv)
#pragma unroll
        for (int r = 0; r < 8; ++r) {
          const int tok = mt * 16 + r + 8 * hlf;
          qbuf[(wv * kL + tok) * kD + dv * 16 + sel] = (h16)oacc[dv][r];
        }
    }
  }
  __syncthreads();

  // ---------------- Phase 3: proj + masked accumulate to global -----------
  // 24 N-tiles; wave w owns {2w, 2w+1}.  K-step k == head index k.
  for (int ntl = 0; ntl < 2; ++ntl) {
    const int nt = wv * 2 + ntl;
    const int j0 = nt * 16;
    v8f acc[4];
#pragma unroll
    for (int mt = 0; mt < 4; ++mt) acc[mt] = zero8();

    for (int k = 0; k < 12; ++k) {
      const v16h bf = *(const v16h*)(wprojT + (size_t)(j0 + sel) * kC + k * 32 + 16 * hlf);
      const h16* abase = qbuf + (k * kL) * kD;   // head-k slab holds channels [32k,32k+32)
#pragma unroll
      for (int mt = 0; mt < 4; ++mt) {
        const h16* ap = abase + (mt * 16 + sel) * kD;
        const v16h af = mkA(ap + 16 * hlf, ap + 16 * hlf + 8);
        acc[mt] = wmma_f16(af, bf, acc[mt]);
      }
    }

    const float bj = b_proj[j0 + sel];
#pragma unroll
    for (int mt = 0; mt < 4; ++mt) {
#pragma unroll
      for (int r = 0; r < 8; ++r) {
        const int t  = mt * 16 + r + 8 * hlf;
        const int hh = whi * 8 + (t >> 3) - sh;
        const int ww = wwi * 8 + (t & 7) - sw;
        // crop to [0,H)x[0,W) and apply shift mask (rows<sh / cols<sw drop out)
        if (hh >= sh && hh < kH && ww >= sw && ww < kW) {
          const size_t o = (((size_t)b * kH + hh) * kW + ww) * kC + j0 + sel;
          const float v = acc[mt][r] + bj;
          if (first) out[o] = v;
          else       out[o] += v;
        }
      }
    }
  }
}

// Transpose + f32->f16 convert both weight matrices into workspace.
__global__ void prep_weights_kernel(const float* __restrict__ wqkv,
                                    const float* __restrict__ wproj,
                                    h16* __restrict__ wqkvT,
                                    h16* __restrict__ wprojT) {
  const int i = blockIdx.x * blockDim.x + threadIdx.x;
  if (i < k3C * kC) {          // wqkvT[j][c] = wqkv[c][j]
    const int j = i / kC, c = i - j * kC;
    wqkvT[i] = (h16)wqkv[(size_t)c * k3C + j];
  }
  if (i < kC * kC) {           // wprojT[j][c] = wproj[c][j]
    const int j = i / kC, c = i - j * kC;
    wprojT[i] = (h16)wproj[(size_t)c * kC + j];
  }
}

// out /= (count + 1e-8), count(h,w) = ch(h)*cw(w) analytically.
__global__ void normalize_kernel(float* __restrict__ out) {
  const size_t total = (size_t)2 * kH * kW * kC;
  const size_t i = (size_t)blockIdx.x * blockDim.x + threadIdx.x;
  if (i >= total) return;
  const size_t hw = i / kC;
  const int w = (int)(hw % kW);
  const int h = (int)((hw / kW) % kH);
  const float ch = 1.f + (float)(h >= 4) + (float)(h >= 8) + (float)(h >= 12);
  const float cw = 1.f + (float)(w >= 4);
  out[i] = out[i] / (ch * cw + 1e-8f);
}

extern "C" void kernel_launch(void* const* d_in, const int* in_sizes, int n_in,
                              void* d_out, int out_size, void* d_ws, size_t ws_size,
                              hipStream_t stream) {
  const float* x      = (const float*)d_in[0];
  const float* w_qkv  = (const float*)d_in[1];
  const float* b_qkv  = (const float*)d_in[2];
  const float* w_proj = (const float*)d_in[3];
  const float* b_proj = (const float*)d_in[4];
  float* out = (float*)d_out;

  h16* wqkvT  = (h16*)d_ws;                 // 1152*384 halfs
  h16* wprojT = wqkvT + (size_t)k3C * kC;   //  384*384 halfs

  {
    const int n = k3C * kC;
    prep_weights_kernel<<<(n + 255) / 256, 256, 0, stream>>>(w_qkv, w_proj, wqkvT, wprojT);
  }

  for (int i = 0; i < 8; ++i) {
    const int sh = (i / 2) * 4;
    const int sw = (i % 2) * 4;
    const int ph = (8 - (kH + sh) % 8) % 8;
    const int pw = (8 - (kW + sw) % 8) % 8;
    const int nh = (kH + sh + ph) / 8;
    const int nw = (kW + sw + pw) / 8;
    const int blocks = 2 * nh * nw;
    win_msa_kernel<<<blocks, 384, 0, stream>>>(x, b_qkv, b_proj, wqkvT, wprojT,
                                               out, sh, sw, nh, nw, (i == 0) ? 1 : 0);
  }

  {
    const size_t total = (size_t)2 * kH * kW * kC;
    normalize_kernel<<<(int)((total + 255) / 256), 256, 0, stream>>>(out);
  }
}
